// HypoNet_47399259079160
// MI455X (gfx1250) — matrix operand order repaired and checked by reading, not compiled, not run on hardware
//
#include <hip/hip_runtime.h>

typedef _Float16 v16h __attribute__((ext_vector_type(16)));
typedef _Float16 v8h  __attribute__((ext_vector_type(8)));
typedef float    v8f  __attribute__((ext_vector_type(8)));

#define NBATCH 4
#define IMG_H 256
#define IMG_W 256
#define LAT_H 64
#define LAT_W 64
#define C_LAT 32
#define NPIX  (IMG_H * IMG_W)

// staging row: 128 features + 8 halfs pad -> 272B rows (16B aligned, bank-skewed)
#define ROWH 136
#define WAVE_STAGE (16 * ROWH)

// -------- workspace layout (bytes) --------
// Packed f16 A-fragments, slab-major: frag index = kt*8 + mt, 1024B per frag,
// every layer padded to 8 M-tiles -> every K-step slab is exactly 8KB.
//   L0: 2 slabs = 16KB ; L1..L4: 4 slabs = 32KB each
#define SLAB_BYTES 8192
#define PACK_OFF_L0 0
#define PACK_OFF_L1 16384
#define PACK_OFF_L2 49152
#define PACK_OFF_L3 81920
#define PACK_OFF_L4 114688
#define PACK_BATCH_STRIDE 147456
#define BIAS_BASE (PACK_BATCH_STRIDE * NBATCH)  // 589824, + li*512 each
// total ws needed: 589824 + 5*512 = 592384 bytes

// =====================================================================
// Phase 1: modulate, normalize (over fan-in), pack weights into the
// 16-bit WMMA A-matrix per-lane layout (slab-major); write f32 bias rows.
// grid = NBATCH*5 blocks, 256 threads
// =====================================================================
__global__ __launch_bounds__(256) void pack_weights_kernel(
    const float* __restrict__ w0, const float* __restrict__ w1,
    const float* __restrict__ w2, const float* __restrict__ w3,
    const float* __restrict__ w4,
    const float* __restrict__ m0, const float* __restrict__ m1,
    const float* __restrict__ m2, const float* __restrict__ m3,
    const float* __restrict__ m4,
    char* __restrict__ ws)
{
    const int blk = blockIdx.x;
    const int b  = blk / 5;
    const int li = blk % 5;

    const float* wTab[5] = {w0, w1, w2, w3, w4};
    const float* mTab[5] = {m0, m1, m2, m3, m4};
    const int KrealT[5]   = {35, 128, 128, 128, 128};
    const int KtT[5]      = {2, 4, 4, 4, 4};
    const int OT[5]       = {128, 128, 128, 128, 2};
    const int packOffT[5] = {PACK_OFF_L0, PACK_OFF_L1, PACK_OFF_L2,
                             PACK_OFF_L3, PACK_OFF_L4};

    const int Kreal = KrealT[li];
    const int O     = OT[li];
    const int Kt    = KtT[li];

    const float* wp = wTab[li];
    const float* mp = mTab[li] + (size_t)b * Kreal * O;

    __shared__ float invn[128];

    for (int o = threadIdx.x; o < O; o += blockDim.x) {
        float s = 0.0f;
        for (int k = 0; k < Kreal; ++k) {
            float v = wp[k * O + o] * mp[k * O + o];
            s += v * v;
        }
        invn[o] = 1.0f / fmaxf(sqrtf(s), 1e-12f);
    }
    __syncthreads();

    unsigned* outp =
        (unsigned*)(ws + (size_t)b * PACK_BATCH_STRIDE + packOffT[li]);
    const int total = Kt * 8 * 256;  // dwords (slab = 8 frags x 256 dwords)
    for (int d = threadIdx.x; d < total; d += blockDim.x) {
        int frag = d >> 8;          // kt*8 + mt
        int kt   = frag >> 3;
        int mt   = frag & 7;
        int lane = (d >> 3) & 31;
        int v    = d & 7;
        int o  = mt * 16 + (lane & 15);
        int hi = lane >> 4;
        int kin = (v < 4) ? (hi * 8 + 2 * v) : (16 + hi * 8 + 2 * (v - 4));
        int k0  = kt * 32 + kin;
        _Float16 h0 = (_Float16)0.f, h1 = (_Float16)0.f;
        if (o < O) {
            if (k0 < Kreal)
                h0 = (_Float16)(wp[k0 * O + o] * mp[k0 * O + o] * invn[o]);
            if (k0 + 1 < Kreal)
                h1 = (_Float16)(wp[(k0 + 1) * O + o] * mp[(k0 + 1) * O + o] * invn[o]);
        }
        union { _Float16 h[2]; unsigned u; } pk;
        pk.h[0] = h0; pk.h[1] = h1;
        outp[d] = pk.u;
    }

    if (b == 0) {  // bias rows are batch-independent (last row of w, unmodulated)
        float* bo = (float*)(ws + BIAS_BASE + li * 512);
        const int Opad = (li == 4) ? 16 : 128;
        for (int o = threadIdx.x; o < Opad; o += blockDim.x)
            bo[o] = (o < O) ? wp[Kreal * O + o] : 0.0f;
    }
}

// =====================================================================
// Async global -> LDS copy helpers (CDNA5 ASYNCcnt path)
// =====================================================================
__device__ __forceinline__ void async_ld16(const void* g, void* l)
{
    asm volatile("global_load_async_to_lds_b128 %0, %1, off"
                 :
                 : "v"((unsigned)(uintptr_t)l),
                   "v"((unsigned long long)(uintptr_t)g)
                 : "memory");
}

template <int N>
__device__ __forceinline__ void wait_asynccnt()
{
#if __has_builtin(__builtin_amdgcn_s_wait_asynccnt)
    __builtin_amdgcn_s_wait_asynccnt(N);
#else
    asm volatile("s_wait_asynccnt %0" : : "i"(N) : "memory");
#endif
}

// copy one 8KB slab with 256 threads: 2 async b128 per thread (2 per wave)
__device__ __forceinline__ void copy_slab(const char* gsrc, char* ldst, int tid)
{
    const int off = tid * 16;
    async_ld16(gsrc + off,        ldst + off);
    async_ld16(gsrc + off + 4096, ldst + off + 4096);
}

// =====================================================================
// Phase 2: per-pixel MLP via WMMA f32_16x16x32_f16.
// hidden^T = W'^T (A, LDS double-buffered slabs) x X^T (B, LDS staging).
// =====================================================================
template <int KT, int MT, bool DO_SIN>
__device__ __forceinline__ void run_layer(const char* packBase,
                                          const float* __restrict__ bias,
                                          _Float16* st, char* wslab,
                                          int tid, int lane, v8f* acc)
{
    const int hi = lane >> 4;
    const int ln = lane & 15;

    copy_slab(packBase, wslab, tid);  // prefetch slab 0 -> buf0

    // All B fragments up front: staging is immutable for the whole layer
    // (previous layer's stores are behind a barrier).
    v16h bfr[KT];
#pragma unroll
    for (int kt = 0; kt < KT; ++kt) {
        const _Float16* bp = st + ln * ROWH + kt * 32 + hi * 8;
        v8h blo = *(const v8h*)bp;
        v8h bh2 = *(const v8h*)(bp + 16);
        bfr[kt] = __builtin_shufflevector(
            blo, bh2, 0, 1, 2, 3, 4, 5, 6, 7, 8, 9, 10, 11, 12, 13, 14, 15);
    }

#pragma unroll
    for (int mt = 0; mt < MT; ++mt)
        acc[mt] = *(const v8f*)(bias + mt * 16 + hi * 8);

#pragma unroll
    for (int kt = 0; kt < KT; ++kt) {
        if (kt + 1 < KT) {
            copy_slab(packBase + (kt + 1) * SLAB_BYTES,
                      wslab + ((kt + 1) & 1) * SLAB_BYTES, tid);
            wait_asynccnt<2>();   // my 2 copies of slab kt have landed
        } else {
            wait_asynccnt<0>();
        }
        __syncthreads();          // everyone's slab-kt copies have landed

        const char* wb = wslab + (kt & 1) * SLAB_BYTES;

        // software-pipeline A fragments: load mt+1 before WMMA(mt)
        v16h a_cur = *(const v16h*)(wb + lane * 32);
#pragma unroll
        for (int mt = 0; mt < MT; ++mt) {
            v16h a_nxt = a_cur;
            if (mt + 1 < MT)
                a_nxt = *(const v16h*)(wb + (mt + 1) * 1024 + lane * 32);
            acc[mt] = __builtin_amdgcn_wmma_f32_16x16x32_f16(
                false, a_cur, false, bfr[kt], (short)0, acc[mt], false, false);
            a_cur = a_nxt;
        }
        __syncthreads();          // all reads of this buffer done before reuse
    }

    if (DO_SIN) {
#pragma unroll
        for (int mt = 0; mt < MT; ++mt) {
            v8h h;
#pragma unroll
            for (int r = 0; r < 8; ++r)
                h[r] = (_Float16)__sinf(acc[mt][r]);  // W0_SINE = 1.0
            *(v8h*)(st + ln * ROWH + mt * 16 + hi * 8) = h;
        }
        __syncthreads();
    }
}

__global__ __launch_bounds__(256) void hyponet_main_kernel(
    const float* __restrict__ coord, const float* __restrict__ latent,
    const char* __restrict__ ws, float* __restrict__ out)
{
    __shared__ __attribute__((aligned(16))) _Float16 stage[8 * WAVE_STAGE];
    __shared__ __attribute__((aligned(16))) char wslab[2 * SLAB_BYTES];

    const int tid  = threadIdx.x;
    const int lane = tid & 31;
    const int wave = tid >> 5;
    const int b    = blockIdx.x >> 9;                    // 512 blocks per batch
    const int p0   = ((blockIdx.x & 511) * 128) + wave * 16;
    _Float16* st   = &stage[wave * WAVE_STAGE];

    // ---- stage layer-0 inputs: 32 bilinear latent channels + 3 coords ----
    {
        const int pi = lane >> 1;
        const int half = lane & 1;
        const int P = p0 + pi;
        _Float16* row = st + pi * ROWH;
        if (half == 0) {
            const int y = P >> 8, x = P & 255;
            float sy = (y + 0.5f) * 0.25f - 0.5f;
            float sx = (x + 0.5f) * 0.25f - 0.5f;
            int y0 = (int)floorf(sy), x0 = (int)floorf(sx);
            float fy = sy - (float)y0, fx = sx - (float)x0;
            int y0c = min(max(y0, 0), LAT_H - 1);
            int y1c = min(max(y0 + 1, 0), LAT_H - 1);
            int x0c = min(max(x0, 0), LAT_W - 1);
            int x1c = min(max(x0 + 1, 0), LAT_W - 1);
            float w00 = (1.f - fy) * (1.f - fx), w01 = (1.f - fy) * fx;
            float w10 = fy * (1.f - fx),         w11 = fy * fx;
            const float* Lb = latent + (size_t)b * LAT_H * LAT_W * C_LAT;
            const float4* q00 = (const float4*)(Lb + (y0c * LAT_W + x0c) * C_LAT);
            const float4* q01 = (const float4*)(Lb + (y0c * LAT_W + x1c) * C_LAT);
            const float4* q10 = (const float4*)(Lb + (y1c * LAT_W + x0c) * C_LAT);
            const float4* q11 = (const float4*)(Lb + (y1c * LAT_W + x1c) * C_LAT);
#pragma unroll
            for (int q = 0; q < 8; ++q) {
                float4 a = q00[q], c = q01[q], d = q10[q], e = q11[q];
                row[q * 4 + 0] = (_Float16)(w00 * a.x + w01 * c.x + w10 * d.x + w11 * e.x);
                row[q * 4 + 1] = (_Float16)(w00 * a.y + w01 * c.y + w10 * d.y + w11 * e.y);
                row[q * 4 + 2] = (_Float16)(w00 * a.z + w01 * c.z + w10 * d.z + w11 * e.z);
                row[q * 4 + 3] = (_Float16)(w00 * a.w + w01 * c.w + w10 * d.w + w11 * e.w);
            }
        } else {
            const float* cp = coord + ((size_t)b * NPIX + P) * 3;
            row[32] = (_Float16)cp[0];
            row[33] = (_Float16)cp[1];
            row[34] = (_Float16)cp[2];
#pragma unroll
            for (int f = 35; f < 64; ++f) row[f] = (_Float16)0.f;
        }
    }
    __syncthreads();

    const char* packBatch = ws + (size_t)b * PACK_BATCH_STRIDE;
    const float* biasBase = (const float*)(ws + BIAS_BASE);
    v8f acc[8];

    run_layer<2, 8, true >(packBatch + PACK_OFF_L0, biasBase + 0 * 128, st, wslab, tid, lane, acc);
    run_layer<4, 8, true >(packBatch + PACK_OFF_L1, biasBase + 1 * 128, st, wslab, tid, lane, acc);
    run_layer<4, 8, true >(packBatch + PACK_OFF_L2, biasBase + 2 * 128, st, wslab, tid, lane, acc);
    run_layer<4, 8, true >(packBatch + PACK_OFF_L3, biasBase + 3 * 128, st, wslab, tid, lane, acc);
    run_layer<4, 1, false>(packBatch + PACK_OFF_L4, biasBase + 4 * 128, st, wslab, tid, lane, acc);

    // D layout layer-4: lanes 0..15 hold pixel N=lane; acc[0][0]=o0, acc[0][1]=o1
    if ((lane >> 4) == 0) {
        const int P = p0 + (lane & 15);
        float2 o2;
        o2.x = acc[0][0] + 0.5f;  // OUTPUT_BIAS
        o2.y = acc[0][1] + 0.5f;
        *(float2*)(out + ((size_t)b * NPIX + P) * 2) = o2;
    }
}

// =====================================================================
extern "C" void kernel_launch(void* const* d_in, const int* in_sizes, int n_in,
                              void* d_out, int out_size, void* d_ws, size_t ws_size,
                              hipStream_t stream)
{
    const float* coord  = (const float*)d_in[0];
    const float* latent = (const float*)d_in[1];
    const float* w0 = (const float*)d_in[2];
    const float* w1 = (const float*)d_in[3];
    const float* w2 = (const float*)d_in[4];
    const float* w3 = (const float*)d_in[5];
    const float* w4 = (const float*)d_in[6];
    const float* m0 = (const float*)d_in[7];
    const float* m1 = (const float*)d_in[8];
    const float* m2 = (const float*)d_in[9];
    const float* m3 = (const float*)d_in[10];
    const float* m4 = (const float*)d_in[11];

    char* ws = (char*)d_ws;

    pack_weights_kernel<<<NBATCH * 5, 256, 0, stream>>>(
        w0, w1, w2, w3, w4, m0, m1, m2, m3, m4, ws);

    // 512 blocks per batch (128 pixels/block, 16 per wave, 8 waves)
    hyponet_main_kernel<<<NBATCH * 512, 256, 0, stream>>>(
        coord, latent, ws, (float*)d_out);
}
